// MagFaceLoss_50414326120960
// MI455X (gfx1250) — matrix operand order repaired and checked by reading, not compiled, not run on hardware
//
#include <hip/hip_runtime.h>

// MagFace loss for MI455X (gfx1250, wave32).
// bn_stats -> row_norm(xn f16, margins, loss_g)
//   -> fused GEMM (f16 WMMA) + weight-norm + margin + scale + fixed-shift
//      softmax partial sums (logits bounded by 64, so shift=64 is exact LSE)
//   -> tiny final loss kernel.

typedef __attribute__((ext_vector_type(16))) _Float16 v16h;
typedef __attribute__((ext_vector_type(2)))  _Float16 v2h;
typedef __attribute__((ext_vector_type(8)))  float    v8f;

#define B_ROWS 256
#define D_DIM  512
#define C_CLS  100000
#define BT_LD  40      // padded LDS row stride in halves (80B, 16B-aligned rows)

// ---------------------------------------------------------------------------
// Kernel 1: BatchNorm column stats (2 blocks x 256 threads, one column each).
// Block 0 also zero-initializes loss accumulators and per-row softmax sums.
// ---------------------------------------------------------------------------
__global__ void bn_stats_kernel(const float* __restrict__ x,
                                const float* __restrict__ gamma,
                                const float* __restrict__ beta,
                                float* __restrict__ scale,
                                float* __restrict__ shift,
                                float* __restrict__ lacc,
                                float* __restrict__ rowsum) {
    const int d = blockIdx.x * 256 + threadIdx.x;
    if (blockIdx.x == 0) {
        rowsum[threadIdx.x] = 0.f;
        if (threadIdx.x == 0) lacc[0] = 0.f;
    }
    float s = 0.f, s2 = 0.f;
    for (int b = 0; b < B_ROWS; ++b) {
        float v = x[b * D_DIM + d];
        s += v; s2 += v * v;
    }
    const float mean = s * (1.f / B_ROWS);
    const float var  = s2 * (1.f / B_ROWS) - mean * mean;   // biased batch var
    const float sc   = gamma[d] * rsqrtf(var + 2e-5f);
    scale[d] = sc;
    shift[d] = beta[d] - mean * sc;
}

// ---------------------------------------------------------------------------
// Kernel 2: per-row ||x_bn||, adaptive margin, x_n in f16, loss_g partial.
// ---------------------------------------------------------------------------
__global__ void row_norm_kernel(const float* __restrict__ x,
                                const float* __restrict__ scale,
                                const float* __restrict__ shift,
                                _Float16* __restrict__ xn,
                                float* __restrict__ cosm,
                                float* __restrict__ sinm,
                                float* __restrict__ lacc) {
    __shared__ float red[256];
    __shared__ float binv;
    const int b = blockIdx.x, t = threadIdx.x;
    const int d0 = t, d1 = t + 256;
    const float xb0 = x[b * D_DIM + d0] * scale[d0] + shift[d0];
    const float xb1 = x[b * D_DIM + d1] * scale[d1] + shift[d1];
    red[t] = xb0 * xb0 + xb1 * xb1;
    __syncthreads();
    for (int off = 128; off; off >>= 1) {
        if (t < off) red[t] += red[t + off];
        __syncthreads();
    }
    if (t == 0) {
        const float nrm = sqrtf(red[0]);
        binv = 1.f / nrm;                              // unclipped norm for x_n
        const float xc  = fminf(fmaxf(nrm, 10.f), 110.f);
        const float ada = (0.8f - 0.45f) / (110.f - 10.f) * (xc - 10.f) + 0.45f;
        cosm[b] = cosf(ada);
        sinm[b] = sinf(ada);
        atomicAdd(&lacc[0], xc * (1.f / (110.f * 110.f)) + 1.f / xc);
    }
    __syncthreads();
    const float inv = binv;
    xn[b * D_DIM + d0] = (_Float16)(xb0 * inv);
    xn[b * D_DIM + d1] = (_Float16)(xb1 * inv);
}

// ---------------------------------------------------------------------------
// Kernel 3: fused cosine GEMM (f16 WMMA, f32 acc) + weight-norm + margin +
// scale + fixed-shift softmax partials. 3125 blocks x 256 threads;
// block = 32 classes x all 256 rows => weight is streamed exactly once.
// ---------------------------------------------------------------------------
__global__ __launch_bounds__(256) void gemm_margin_kernel(
    const float* __restrict__ w,         // [D, C] row-major f32
    const _Float16* __restrict__ xn,     // [B, D] f16
    const float* __restrict__ cosm,
    const float* __restrict__ sinm,
    const int* __restrict__ tgt,
    float* __restrict__ outm,            // [B, C] f32
    float* __restrict__ rowsum) {        // [B]  sum_c exp(out - 64)
    __shared__ __align__(16) _Float16 bt[32][BT_LD];   // weight tile, [n][k] padded
    __shared__ float wsq[32];
    __shared__ float cosm_s[B_ROWS], sinm_s[B_ROWS];
    __shared__ int   tgt_s[B_ROWS];

    const int tid  = threadIdx.x;
    const int lane = tid & 31;
    const int wave = tid >> 5;           // 0..7
    const int n0   = blockIdx.x * 32;

    cosm_s[tid] = cosm[tid];
    sinm_s[tid] = sinm[tid];
    tgt_s[tid]  = tgt[tid];
    if (tid < 32) wsq[tid] = 0.f;

    // Cooperative weight-tile mapping: thread -> (2 k rows, 2 n columns).
    const int k2 = (tid >> 4) * 2;       // 0,2,...,30
    const int n2 = (tid & 15) * 2;       // 0,2,...,30
    float sq0 = 0.f, sq1 = 0.f;          // sumsq for columns n2, n2+1

    // Wave tiling: 4 waves along M (64 rows each), 2 waves along N (16 cols).
    const int mw  = wave & 3;
    const int nw  = wave >> 2;
    const int l15 = lane & 15;
    const int hi  = lane >> 4;           // lane half selects WMMA K sub-range
    const int nl  = l15 + nw * 16;       // local column 0..31

    v8f acc0 = {}, acc1 = {}, acc2 = {}, acc3 = {};

    for (int kt = 0; kt < 16; ++kt) {
        const float* wp0 = w + (size_t)(kt * 32 + k2) * C_CLS + n0 + n2;
        const float* wp1 = wp0 + C_CLS;
        const float2 a0 = *(const float2*)wp0;   // (k2  , n2), (k2  , n2+1)
        const float2 a1 = *(const float2*)wp1;   // (k2+1, n2), (k2+1, n2+1)
        sq0 += a0.x * a0.x + a1.x * a1.x;
        sq1 += a0.y * a0.y + a1.y * a1.y;

        // Stream-prefetch two K-tiles ahead (1 lane per 128B segment).
        if ((tid & 15) == 0) {
            __builtin_prefetch(wp0 + (size_t)64 * C_CLS, 0, 0);
            __builtin_prefetch(wp1 + (size_t)64 * C_CLS, 0, 0);
        }

        __syncthreads();                 // previous tile fully consumed
        *(v2h*)&bt[n2 + 0][k2] = (v2h){(_Float16)a0.x, (_Float16)a1.x};
        *(v2h*)&bt[n2 + 1][k2] = (v2h){(_Float16)a0.y, (_Float16)a1.y};
        __syncthreads();

        // B fragment: lane column nl; lanes 0-15 -> K 0..15, 16-31 -> K 16..31.
        union { v16h v; uint4 u[2]; } bf;
        const uint4* bp = (const uint4*)&bt[nl][hi * 16];
        bf.u[0] = bp[0];
        bf.u[1] = bp[1];

        auto do_tile = [&](int mt, v8f& accv) {
            // A fragment: lanes 0-15 hold K 0..15, lanes 16-31 K 16..31.
            const _Float16* ap =
                xn + (size_t)((mw * 64 + mt * 16 + l15) * D_DIM + kt * 32 + hi * 16);
            union { v16h v; uint4 u[2]; } af;
            af.u[0] = *(const uint4*)ap;
            af.u[1] = *(const uint4*)(ap + 8);
            accv = __builtin_amdgcn_wmma_f32_16x16x32_f16(
                false, af.v, false, bf.v, (short)0, accv, false, false);
        };
        do_tile(0, acc0);
        do_tile(1, acc1);
        do_tile(2, acc2);
        do_tile(3, acc3);
    }

    // Column sum-of-squares reduction (fused weight normalization).
    atomicAdd(&wsq[n2 + 0], sq0);
    atomicAdd(&wsq[n2 + 1], sq1);
    __syncthreads();

    const float rw = rsqrtf(wsq[nl]);
    const int cg = n0 + nl;              // global class id of this lane

    auto epilogue = [&](int mt, const v8f& accv) {
#pragma unroll
        for (int j = 0; j < 8; ++j) {
            const int m = mw * 64 + mt * 16 + j + hi * 8;   // C/D layout row
            float cv = fminf(fmaxf(accv[j] * rw, -1.f), 1.f);
            const float sv = sqrtf(fmaxf(1.f - cv * cv, 0.f));
            const float ctm = (cv > 0.f) ? cv * cosm_s[m] - sv * sinm_s[m] : cv;
            const float val = 64.f * ((tgt_s[m] == cg) ? ctm : cv);
            outm[(size_t)m * C_CLS + cg] = val;
            // Fixed-shift softmax partial: logits bounded by 64, shift exact.
            float p = expf(val - 64.f);
            p += __shfl_xor(p, 8);       // reduce across the 16 columns this
            p += __shfl_xor(p, 4);       // half-wave owns (xor<16 stays in half)
            p += __shfl_xor(p, 2);
            p += __shfl_xor(p, 1);
            if (l15 == 0) atomicAdd(&rowsum[m], p);
        }
    };
    epilogue(0, acc0);
    epilogue(1, acc1);
    epilogue(2, acc2);
    epilogue(3, acc3);
}

// ---------------------------------------------------------------------------
// Kernel 4: final loss. loss_id[b] = log(rowsum[b]) + 64 - out[b, tgt[b]].
// ---------------------------------------------------------------------------
__global__ void final_loss_kernel(const float* __restrict__ outm,
                                  const int* __restrict__ tgt,
                                  const float* __restrict__ rowsum,
                                  const float* __restrict__ lacc,
                                  float* __restrict__ out0) {
    __shared__ float red[256];
    const int t = threadIdx.x;
    const float tv = outm[(size_t)t * C_CLS + tgt[t]];
    red[t] = logf(rowsum[t]) + 64.f - tv;
    __syncthreads();
    for (int off = 128; off; off >>= 1) {
        if (t < off) red[t] += red[t + off];
        __syncthreads();
    }
    if (t == 0)
        out0[0] = red[0] * (1.f / B_ROWS) + 20.f * (lacc[0] * (1.f / B_ROWS));
}

// ---------------------------------------------------------------------------
extern "C" void kernel_launch(void* const* d_in, const int* in_sizes, int n_in,
                              void* d_out, int out_size, void* d_ws, size_t ws_size,
                              hipStream_t stream) {
    (void)in_sizes; (void)n_in; (void)out_size; (void)ws_size;
    const float* x      = (const float*)d_in[0];
    const int*   target = (const int*)d_in[1];
    const float* weight = (const float*)d_in[2];
    const float* gamma  = (const float*)d_in[3];
    const float* beta   = (const float*)d_in[4];

    float* out = (float*)d_out;          // [0] = loss, [1..] = output [256,100000]

    char* ws = (char*)d_ws;
    float* scale  = (float*)ws;          // 512
    float* shift  = scale + 512;         // 512
    float* cosm   = shift + 512;         // 256
    float* sinm   = cosm + 256;          // 256
    float* lacc   = sinm + 256;          // [0] = loss_g sum (+pad)
    float* rowsum = lacc + 4;            // 256 per-row softmax sums
    _Float16* xn = (_Float16*)(ws + 8192);   // 256*512 f16 = 256 KB

    bn_stats_kernel<<<2, 256, 0, stream>>>(x, gamma, beta, scale, shift, lacc, rowsum);
    row_norm_kernel<<<B_ROWS, 256, 0, stream>>>(x, scale, shift, xn, cosm, sinm, lacc);
    gemm_margin_kernel<<<C_CLS / 32, 256, 0, stream>>>(weight, xn, cosm, sinm, target,
                                                       out + 1, rowsum);
    final_loss_kernel<<<1, 256, 0, stream>>>(out + 1, target, rowsum, lacc, out);
}
